// informed_net_68126771249663
// MI455X (gfx1250) — matrix operand order repaired and checked by reading, not compiled, not run on hardware
//
#include <hip/hip_runtime.h>
#include <hip/hip_bf16.h>

typedef __attribute__((ext_vector_type(16))) __bf16 v16bf;
typedef __attribute__((ext_vector_type(8)))  __bf16 v8bf;
typedef __attribute__((ext_vector_type(8)))  float  v8f;
typedef __attribute__((ext_vector_type(4)))  float  v4f;
typedef int v4i_vs __attribute__((vector_size(16)));   // matches builtin param type

#if defined(__HIP_DEVICE_COMPILE__) && \
    __has_builtin(__builtin_amdgcn_global_load_async_to_lds_b128) && \
    __has_builtin(__builtin_amdgcn_s_wait_asynccnt)
#define USE_ASYNC 1
#else
#define USE_ASYNC 0
#endif

static constexpr int TBM = 128;   // block tile M
static constexpr int TBN = 128;   // block tile N
static constexpr int TBK = 32;    // K step == WMMA K for bf16
static constexpr int LSTR = 40;   // 32 + 8 pad -> 80B rows, keeps 16B alignment

union AFrag { v16bf v; v8bf h[2]; };
union P2    { __bf16 b[2]; unsigned u; };

// ---------------------------------------------------------------------------
// gather + fp32->bf16 convert: xg[b][j] = bf16(x[b*TOT + ind[j]]),  D = 4096
// ---------------------------------------------------------------------------
__global__ __launch_bounds__(256) void gather_bf16(
    const float* __restrict__ x, const int* __restrict__ ind,
    __bf16* __restrict__ xg)
{
  constexpr int D = 4096, TOT = 16384;
  const int gid = blockIdx.x * 256 + threadIdx.x;
  const int row = gid >> 9;             // D/8 = 512 groups of 8 per row
  const int col = (gid & 511) * 8;
  const float* xr = x + (size_t)row * TOT;
  v8bf o;
#pragma unroll
  for (int i = 0; i < 8; ++i) o[i] = (__bf16)xr[ind[col + i]];
  *(v8bf*)&xg[(size_t)row * D + col] = o;
}

// ---------------------------------------------------------------------------
// D = ReLU(A @ W + bias), A bf16, W/bias fp32 (converted while staging)
// MODE 0: D bf16 (ldd, pointer may carry a column offset)
// MODE 2: D fp32, columns scattered through sind (ldd = row stride)
// Block: 256 threads = 8 waves; tile 128x128x32; wave tile 32x64 (2x4 WMMAs).
// LDS double-buffered, A tile via async global->LDS DMA when available.
// ---------------------------------------------------------------------------
template<int MODE>
__global__ __launch_bounds__(256) void gemm_bias_relu(
    const __bf16* __restrict__ A,
    const float* __restrict__ W, const float* __restrict__ bias,
    __bf16* __restrict__ Dbf, float* __restrict__ Dfp,
    const int* __restrict__ sind,
    int K, int lda, int ldb, int ldd)
{
  __shared__ __bf16 Al[2][TBM * LSTR];  // Al[buf][m][k]
  __shared__ __bf16 Bl[2][TBN * LSTR];  // Bl[buf][n][k] (K-major -> b128 frag reads)

  const int t    = threadIdx.x;
  const int lane = t & 31;
  const int wid  = t >> 5;              // 8 waves
  const int wm   = wid >> 1;            // 0..3 -> 32 rows each
  const int wn   = wid & 1;             // 0..1 -> 64 cols each
  const int tm   = blockIdx.y * TBM;
  const int tn   = blockIdx.x * TBN;

  v8f acc[2][4];
  const v8f vzero = {0.f, 0.f, 0.f, 0.f, 0.f, 0.f, 0.f, 0.f};
#pragma unroll
  for (int mi = 0; mi < 2; ++mi)
#pragma unroll
    for (int ni = 0; ni < 4; ++ni) acc[mi][ni] = vzero;

  // ---- tile-load assignments (256 threads) ----
  const int arow = t >> 1;              // 0..127
  const int acol = (t & 1) * 16;        // 0 or 16
  const int bk0  = (t >> 4) * 2;        // 0,2,...,30   (2 K-rows per thread)
  const int bn0  = (t & 15) * 8;        // 0..120       (8 N-cols per thread)

  const __bf16* aptr = A + (size_t)(tm + arow) * (size_t)lda + acol;
  const float*  wptr = W + (size_t)bk0 * (size_t)ldb + tn + bn0;

  v4f bf4[4];                           // staged B (2 K-rows x 8 N-cols fp32)
  auto loadB = [&](int kt) {
    const float* wr = wptr + (size_t)kt * (size_t)ldb;
    bf4[0] = *(const v4f*)(wr);
    bf4[1] = *(const v4f*)(wr + 4);
    bf4[2] = *(const v4f*)(wr + ldb);
    bf4[3] = *(const v4f*)(wr + ldb + 4);
  };
  auto storeB = [&](int buf) {
#pragma unroll
    for (int j = 0; j < 8; ++j) {       // pack K-adjacent pair -> one b32 store
      P2 p;
      p.b[0] = (__bf16)((j < 4) ? bf4[0][j] : bf4[1][j - 4]);   // k = bk0
      p.b[1] = (__bf16)((j < 4) ? bf4[2][j] : bf4[3][j - 4]);   // k = bk0+1
      *(unsigned*)&Bl[buf][(bn0 + j) * LSTR + bk0] = p.u;
    }
  };

#if USE_ASYNC
  // A tile: per-lane async DMA, 32B per thread as two b128 (the instruction
  // offset applies to BOTH the global and LDS addresses, so one base pointer
  // pair serves both halves with offset 0 / 16)
  auto issueA = [&](int kt, int buf) {
    v4i_vs* ar = (v4i_vs*)(aptr + kt);
    v4i_vs* ld = (v4i_vs*)&Al[buf][arow * LSTR + acol];
    __builtin_amdgcn_global_load_async_to_lds_b128(ar, ld, 0, 0);
    __builtin_amdgcn_global_load_async_to_lds_b128(ar, ld, 16, 0);
  };
#else
  v8bf ap0, ap1;                        // staged A (16 bf16)
  auto loadA = [&](int kt) {
    const __bf16* ar = aptr + kt;
    ap0 = *(const v8bf*)(ar);
    ap1 = *(const v8bf*)(ar + 8);
  };
  auto storeA = [&](int buf) {
    *(v8bf*)&Al[buf][arow * LSTR + acol]     = ap0;
    *(v8bf*)&Al[buf][arow * LSTR + acol + 8] = ap1;
  };
#endif

  // ---- prologue: stage tile 0 ----
#if USE_ASYNC
  issueA(0, 0);
#else
  loadA(0);
  storeA(0);
#endif
  loadB(0);
  storeB(0);

  const int am   = wm * 32 + (lane & 15);
  const int aklo = (lane < 16) ? 0 : 8;   // A frag: K 0-7/16-23 vs 8-15/24-31
  const int bn   = wn * 64 + (lane & 15);
  const int bkb  = (lane < 16) ? 0 : 8;   // B frag: K 0-7/16-23 vs 8-15/24-31

  int cur = 0;
  for (int kt = 0; kt < K; kt += TBK) {
#if USE_ASYNC
    __builtin_amdgcn_s_wait_asynccnt(0);  // my DMA writes into buf(cur) done
#endif
    __syncthreads();                      // buf(cur) published; reads of buf(cur^1) done
    const bool more = (kt + TBK) < K;
    if (more) {
#if USE_ASYNC
      issueA(kt + TBK, cur ^ 1);          // DMA next A tile while WMMAs run
#else
      loadA(kt + TBK);
#endif
      loadB(kt + TBK);
    }

    AFrag a[2], b[4];
#pragma unroll
    for (int mi = 0; mi < 2; ++mi) {
      const __bf16* p = &Al[cur][(am + mi * 16) * LSTR + aklo];
      a[mi].h[0] = *(const v8bf*)(p);
      a[mi].h[1] = *(const v8bf*)(p + 16);
    }
#pragma unroll
    for (int ni = 0; ni < 4; ++ni) {
      const __bf16* p = &Bl[cur][(bn + ni * 16) * LSTR + bkb];
      b[ni].h[0] = *(const v8bf*)(p);
      b[ni].h[1] = *(const v8bf*)(p + 16);
    }
#pragma unroll
    for (int mi = 0; mi < 2; ++mi)
#pragma unroll
      for (int ni = 0; ni < 4; ++ni)
        acc[mi][ni] = __builtin_amdgcn_wmma_f32_16x16x32_bf16(
            false, a[mi].v, false, b[ni].v, (short)0, acc[mi][ni], false, false);

    if (more) {
      storeB(cur ^ 1);
#if !USE_ASYNC
      storeA(cur ^ 1);
#endif
    }
    cur ^= 1;
  }

  // ---- epilogue: bias + ReLU ; C/D layout: VGPR g -> M = g + 8*(lane>=16), N = lane&15
#pragma unroll
  for (int ni = 0; ni < 4; ++ni) {
    const int ng = tn + wn * 64 + ni * 16 + (lane & 15);
    const float bv = bias[ng];
    const int scol = (MODE == 2) ? sind[ng] : ng;
#pragma unroll
    for (int mi = 0; mi < 2; ++mi) {
      const int rbase = tm + wm * 32 + mi * 16 + ((lane >> 4) << 3);
#pragma unroll
      for (int g = 0; g < 8; ++g) {
        float d = acc[mi][ni][g] + bv;
        d = d > 0.f ? d : 0.f;
        if (MODE == 2) {
          Dfp[(size_t)(rbase + g) * (size_t)ldd + scol] = d;
        } else {
          Dbf[(size_t)(rbase + g) * (size_t)ldd + ng] = (__bf16)d;
        }
      }
    }
  }
}

extern "C" void kernel_launch(void* const* d_in, const int* in_sizes, int n_in,
                              void* d_out, int out_size, void* d_ws, size_t ws_size,
                              hipStream_t stream) {
  (void)in_sizes; (void)n_in; (void)out_size; (void)ws_size;
  // Input order (dict insertion order):
  //  [0] x ; then (W,b) pairs for {epi,met,dia,tot} x {e1,e2,d1,d2} = [1..32];
  //  [33,34] param ; [35,36] re_param ; [37..40] epi/met/dia/tot index arrays.
  const float* x = (const float*)d_in[0];
  auto P = [&](int i) { return (const float*)d_in[i]; };
  const int* ind[4] = { (const int*)d_in[37], (const int*)d_in[38],
                        (const int*)d_in[39], (const int*)d_in[40] };

  constexpr int B = 4096, D = 4096, H = 2048, T = 256, PR = 128, TOT = 16384;

  char* ws = (char*)d_ws;
  __bf16* xg   = (__bf16*)(ws);                        // [B,D]   32 MB (per-region reuse)
  __bf16* h    = (__bf16*)(ws + (size_t)(32u << 20));  // [B,H]   16 MB (reused 8x)
  __bf16* tcat = (__bf16*)(ws + (size_t)(48u << 20));  // [B,4T]   8 MB
  __bf16* z    = (__bf16*)(ws + (size_t)(56u << 20));  // [B,PR]   1 MB
  __bf16* z2   = (__bf16*)(ws + (size_t)(57u << 20));  // [B,4T]   8 MB
  float* out = (float*)d_out;

  dim3 blk(256);
  const int gy = B / TBM;  // 32 row-tiles everywhere

  // ---------------- encode ----------------
  for (int r = 0; r < 4; ++r) {
    const int base = 1 + r * 8;
    // gather region columns of x -> bf16 xg  (each x element read exactly once)
    gather_bf16<<<dim3((B * D / 8) / 256), blk, 0, stream>>>(x, ind[r], xg);
    // e1: [B,D] x [D,H] -> h
    gemm_bias_relu<0><<<dim3(H / TBN, gy), blk, 0, stream>>>(
        xg, P(base + 0), P(base + 1), h, nullptr, nullptr, D, D, H, H);
    // e2: [B,H] x [H,T] -> tcat columns [r*T, (r+1)*T)
    gemm_bias_relu<0><<<dim3(T / TBN, gy), blk, 0, stream>>>(
        h, P(base + 2), P(base + 3), tcat + r * T, nullptr, nullptr, H, H, T, 4 * T);
  }
  // param: [B,4T] x [4T,PR] -> z
  gemm_bias_relu<0><<<dim3(PR / TBN, gy), blk, 0, stream>>>(
      tcat, P(33), P(34), z, nullptr, nullptr, 4 * T, 4 * T, PR, PR);
  // re_param: [B,PR] x [PR,4T] -> z2
  gemm_bias_relu<0><<<dim3((4 * T) / TBN, gy), blk, 0, stream>>>(
      z, P(35), P(36), z2, nullptr, nullptr, PR, PR, 4 * T, 4 * T);
  // ---------------- decode ----------------
  for (int r = 0; r < 4; ++r) {
    const int base = 1 + r * 8;
    // d1: z2 columns [r*T,(r+1)*T) x [T,H] -> h
    gemm_bias_relu<0><<<dim3(H / TBN, gy), blk, 0, stream>>>(
        z2 + r * T, P(base + 4), P(base + 5), h, nullptr, nullptr, T, 4 * T, H, H);
    // d2: [B,H] x [H,D] -> fp32, scattered into d_out via ind[r]
    gemm_bias_relu<2><<<dim3(D / TBN, gy), blk, 0, stream>>>(
        h, P(base + 6), P(base + 7), nullptr, out, ind[r], H, H, D, TOT);
  }
}
